// Decoder_11871289606393
// MI455X (gfx1250) — matrix operand order repaired and checked by reading, not compiled, not run on hardware
//
#include <hip/hip_runtime.h>
#include <hip/hip_bf16.h>

// ---------------- types for CDNA5 WMMA ----------------
typedef __attribute__((ext_vector_type(16))) __bf16 v16bf;
typedef __attribute__((ext_vector_type(8)))  __bf16 v8bf;
typedef __attribute__((ext_vector_type(8)))  float  v8f;

// problem constants (fixed by reference setup)
#define TT 64
#define BB 16
#define SS 400
#define EE 128
#define HH 256
#define C2 512      // 2H
#define DV 50000
#define EXT 50
#define TB (TT*BB)  // 1024

// ---- fast device math: use gfx1250 TRANS ops instead of ocml expansions ----
__device__ __forceinline__ float fast_rcp(float x) {
#if __has_builtin(__builtin_amdgcn_rcpf)
  return __builtin_amdgcn_rcpf(x);
#else
  return 1.0f / x;
#endif
}
__device__ __forceinline__ float fast_tanh(float x) {
#if __has_builtin(__builtin_amdgcn_tanhf)
  return __builtin_amdgcn_tanhf(x);
#else
  return tanhf(x);
#endif
}
__device__ __forceinline__ float fast_sigmoid(float x) {
  return fast_rcp(1.0f + __expf(-x));
}

// A fragment (16x32, row-major source, f32 -> bf16 via native cvt). lane&15 = M.
// slot i : K = 16*hi + i (contiguous 16 elements)
__device__ __forceinline__ v16bf make_a_frag_f32(const float* xp, int hi) {
  float4 x0 = *(const float4*)(xp + 16 * hi);
  float4 x1 = *(const float4*)(xp + 16 * hi + 4);
  float4 x2 = *(const float4*)(xp + 16 * hi + 8);
  float4 x3 = *(const float4*)(xp + 16 * hi + 12);
  v16bf a;
  a[0]  = (__bf16)x0.x; a[1]  = (__bf16)x0.y; a[2]  = (__bf16)x0.z; a[3]  = (__bf16)x0.w;
  a[4]  = (__bf16)x1.x; a[5]  = (__bf16)x1.y; a[6]  = (__bf16)x1.z; a[7]  = (__bf16)x1.w;
  a[8]  = (__bf16)x2.x; a[9]  = (__bf16)x2.y; a[10] = (__bf16)x2.z; a[11] = (__bf16)x2.w;
  a[12] = (__bf16)x3.x; a[13] = (__bf16)x3.y; a[14] = (__bf16)x3.z; a[15] = (__bf16)x3.w;
  return a;
}
// B fragment (32x16). Source is W[N,K] row-major (we compute X @ W^T).
// lane&15 = N row of W ; slot i : K = 16*hi + i (contiguous 16 elements)
__device__ __forceinline__ v16bf make_b_frag_f32(const float* wp, int hi) {
  const float* p = wp + 16 * hi;
  float4 x0 = *(const float4*)(p);
  float4 x1 = *(const float4*)(p + 4);
  float4 x2 = *(const float4*)(p + 8);
  float4 x3 = *(const float4*)(p + 12);
  v16bf b;
  b[0]  = (__bf16)x0.x; b[1]  = (__bf16)x0.y; b[2]  = (__bf16)x0.z; b[3]  = (__bf16)x0.w;
  b[4]  = (__bf16)x1.x; b[5]  = (__bf16)x1.y; b[6]  = (__bf16)x1.z; b[7]  = (__bf16)x1.w;
  b[8]  = (__bf16)x2.x; b[9]  = (__bf16)x2.y; b[10] = (__bf16)x2.z; b[11] = (__bf16)x2.w;
  b[12] = (__bf16)x3.x; b[13] = (__bf16)x3.y; b[14] = (__bf16)x3.z; b[15] = (__bf16)x3.w;
  return b;
}

// ---------------- generic Y = act(X @ W^T + bias) via WMMA ----------------
// X:[M,K] f32, W:[N,K] f32, Y:[M,N] f32. M,N mult of 16, K mult of 32.
// block = 128 (4 waves), wave w -> n_tile = bx*4+w, m_tile = by.
__global__ __launch_bounds__(128)
void gemm_xwT_f32(const float* __restrict__ X, const float* __restrict__ W,
                  const float* __restrict__ bias, float* __restrict__ Y,
                  int M, int N, int K, int act) {
  const int lane = threadIdx.x & 31;
  const int wid  = threadIdx.x >> 5;
  const int n_tile = blockIdx.x * 4 + wid;
  if (n_tile * 16 >= N) return;
  const int m_tile = blockIdx.y;
  const int hi = lane >> 4;
  const int mrow = m_tile * 16 + (lane & 15);
  const int nrow = n_tile * 16 + (lane & 15);
  v8f acc = {};
  for (int k0 = 0; k0 < K; k0 += 32) {
    v16bf a = make_a_frag_f32(X + (size_t)mrow * K + k0, hi);
    v16bf b = make_b_frag_f32(W + (size_t)nrow * K + k0, hi);
    acc = __builtin_amdgcn_wmma_f32_16x16x32_bf16(false, a, false, b,
                                                  (short)0, acc, false, false);
  }
  const int n = n_tile * 16 + (lane & 15);
  const float bv = bias ? bias[n] : 0.0f;
#pragma unroll
  for (int j = 0; j < 8; ++j) {
    int m = m_tile * 16 + hi * 8 + j;
    float v = acc[j] + bv;
    if (act == 1) v = fast_tanh(v);
    else if (act == 2) v = fast_sigmoid(v);
    Y[(size_t)m * N + n] = v;
  }
}

// ---------------- vocab GEMM: logits = Xb @ Wb^T + bias (bf16 in, f32 out) ---
// Xb:[TB,256] bf16, Wb:[DV,256] bf16, Y row stride ldY (=DV+EXT, into d_out).
// Each wave pins one n_tile's B fragments in VGPRs and sweeps all 64 m_tiles,
// so Wb (25.6MB, resident in 192MB L2) is streamed exactly once per wave.
__global__ __launch_bounds__(128)
void gemm_vocab_bf16(const __bf16* __restrict__ Xb, const __bf16* __restrict__ Wb,
                     const float* __restrict__ bias, float* __restrict__ Y,
                     int M, int N, int K, int ldY) {
  const int lane = threadIdx.x & 31;
  const int wid  = threadIdx.x >> 5;
  const int n_tile = blockIdx.x * 4 + wid;
  if (n_tile * 16 >= N) return;
  const int hi = lane >> 4;
  const int nrow = n_tile * 16 + (lane & 15);
  v16bf bfrag[8];
#pragma unroll
  for (int c = 0; c < 8; ++c)
    bfrag[c] = *(const v16bf*)(Wb + (size_t)nrow * K + 32 * c + 16 * hi);
  const int n = n_tile * 16 + (lane & 15);
  const float bv = bias[n];
  for (int mt = 0; mt < M / 16; ++mt) {
    const __bf16* xp = Xb + (size_t)(mt * 16 + (lane & 15)) * K;
    v8f acc = {};
#pragma unroll
    for (int c = 0; c < 8; ++c) {
      v8bf a0 = *(const v8bf*)(xp + 32 * c + 16 * hi);
      v8bf a1 = *(const v8bf*)(xp + 32 * c + 16 * hi + 8);
      v16bf a = __builtin_shufflevector(a0, a1, 0, 1, 2, 3, 4, 5, 6, 7,
                                        8, 9, 10, 11, 12, 13, 14, 15);
      acc = __builtin_amdgcn_wmma_f32_16x16x32_bf16(false, a, false, bfrag[c],
                                                    (short)0, acc, false, false);
    }
#pragma unroll
    for (int j = 0; j < 8; ++j) {
      int m = mt * 16 + hi * 8 + j;
      Y[(size_t)m * ldY + n] = acc[j] + bv;
    }
  }
}

// ---------------- GRU gate combine ----------------
// gx,gh:[B,3H] (r,z,n order), hprev:[B,H], ym:[B], sout:[B,H]
__global__ void gru_combine_kernel(const float* __restrict__ gx, const float* __restrict__ gh,
                                   const float* __restrict__ hprev, const float* __restrict__ ym,
                                   float* __restrict__ sout, float* __restrict__ hs_out) {
  int i = blockIdx.x * blockDim.x + threadIdx.x;
  if (i >= BB * HH) return;
  int b = i / HH, j = i % HH;
  const float* gxr = gx + (size_t)b * 3 * HH;
  const float* ghr = gh + (size_t)b * 3 * HH;
  float r = fast_sigmoid(gxr[j] + ghr[j]);
  float z = fast_sigmoid(gxr[HH + j] + ghr[HH + j]);
  float nn = fast_tanh(gxr[2 * HH + j] + r * ghr[2 * HH + j]);
  float hp = hprev[i];
  float hnew = (1.0f - z) * nn + z * hp;
  float m = ym[b];
  float sv = m * hnew + (1.0f - m) * hp;
  sout[i] = sv;
  if (hs_out) hs_out[i] = sv;
}

// ---------------- attention score (one wave per (s,b) row) ----------------
__global__ __launch_bounds__(128)
void attn_score_kernel(const float* __restrict__ enc_proj, const float* __restrict__ s1W,
                       const float* __restrict__ Wc, const float* __restrict__ Vat,
                       const float* __restrict__ cov, const float* __restrict__ xmask,
                       float* __restrict__ score) {
  int row = blockIdx.x * 4 + (threadIdx.x >> 5);
  if (row >= SS * BB) return;
  int lane = threadIdx.x & 31;
  int s = row / BB, b = row % BB;
  float cv = cov[b * SS + s];
  float acc = 0.0f;
  for (int j = lane; j < C2; j += 32)
    acc += fast_tanh(enc_proj[(size_t)row * C2 + j] + s1W[b * C2 + j] + cv * Wc[j]) * Vat[j];
#pragma unroll
  for (int off = 16; off > 0; off >>= 1) acc += __shfl_xor(acc, off, 32);
  if (lane == 0) {
    float xm = xmask[row];
    score[row] = (xm == 0.0f) ? -1e9f : acc * xm;
  }
}

// ---------------- softmax over S + context + coverage update ----------------
__global__ __launch_bounds__(256)
void attn_ctx_kernel(const float* __restrict__ score, const float* __restrict__ enc,
                     float* __restrict__ cov, float* __restrict__ ctx_t,
                     float* __restrict__ waws_t, float* __restrict__ covs_t) {
  __shared__ float s_attn[SS];
  __shared__ float s_red[256];
  int b = blockIdx.x, tid = threadIdx.x;
  float lmax = -1e30f;
  for (int s = tid; s < SS; s += 256) lmax = fmaxf(lmax, score[s * BB + b]);
  s_red[tid] = lmax; __syncthreads();
  for (int off = 128; off > 0; off >>= 1) {
    if (tid < off) s_red[tid] = fmaxf(s_red[tid], s_red[tid + off]);
    __syncthreads();
  }
  float rmax = s_red[0]; __syncthreads();
  float lsum = 0.0f;
  for (int s = tid; s < SS; s += 256) {
    float e = __expf(score[s * BB + b] - rmax);
    s_attn[s] = e; lsum += e;
  }
  s_red[tid] = lsum; __syncthreads();
  for (int off = 128; off > 0; off >>= 1) {
    if (tid < off) s_red[tid] += s_red[tid + off];
    __syncthreads();
  }
  float inv = fast_rcp(s_red[0]);
  __syncthreads();
  for (int s = tid; s < SS; s += 256) {
    float a = s_attn[s] * inv;
    s_attn[s] = a;
    float c0 = cov[b * SS + s];
    covs_t[b * SS + s] = c0;      // coverage BEFORE update (scan emits carry)
    waws_t[b * SS + s] = a;
    cov[b * SS + s] = c0 + a;
  }
  __syncthreads();
  for (int j = tid; j < C2; j += 256) {
    float acc = 0.0f;
    for (int s = 0; s < SS; ++s) acc += s_attn[s] * enc[((size_t)s * BB + b) * C2 + j];
    ctx_t[b * C2 + j] = acc;
  }
}

// ---------------- pack h = [hs | ctx | emb] ----------------
__global__ void pack_hcat_kernel(const float* __restrict__ hs, const float* __restrict__ ctx,
                                 const float* __restrict__ emb, float* __restrict__ hcat) {
  const int W = HH + C2 + EE;  // 896
  int i = blockIdx.x * blockDim.x + threadIdx.x;
  if (i >= TB * W) return;
  int row = i / W, c = i % W;
  float v;
  if (c < HH)            v = hs[(size_t)row * HH + c];
  else if (c < HH + C2)  v = ctx[(size_t)row * C2 + (c - HH)];
  else                   v = emb[(size_t)row * EE + (c - HH - C2)];
  hcat[i] = v;
}

// ---------------- pointer gate g = sigmoid(h @ Wp^T + b) ----------------
__global__ __launch_bounds__(128)
void gate_kernel(const float* __restrict__ hcat, const float* __restrict__ Wp,
                 const float* __restrict__ Wpb, float* __restrict__ g) {
  const int Kc = HH + C2 + EE;
  int row = blockIdx.x * 4 + (threadIdx.x >> 5);
  if (row >= TB) return;
  int lane = threadIdx.x & 31;
  float acc = 0.0f;
  for (int j = lane; j < Kc; j += 32) acc += hcat[(size_t)row * Kc + j] * Wp[j];
#pragma unroll
  for (int off = 16; off > 0; off >>= 1) acc += __shfl_xor(acc, off, 32);
  if (lane == 0) g[row] = fast_sigmoid(acc + Wpb[0]);
}

// ---------------- f32 -> bf16 ----------------
__global__ void f32_to_bf16_kernel(const float* __restrict__ in, __bf16* __restrict__ out,
                                   size_t n) {
  size_t i = (size_t)blockIdx.x * blockDim.x + threadIdx.x;
  if (i < n) out[i] = (__bf16)in[i];
}

// ------------- in-place vocab softmax, scale by g, zero ext tail -------------
__global__ __launch_bounds__(256)
void vocab_softmax_kernel(float* __restrict__ Y, const float* __restrict__ g) {
  int row = blockIdx.x, tid = threadIdx.x;
  float* yr = Y + (size_t)row * (DV + EXT);
  __shared__ float s_red[256];
  float lmax = -1e30f;
  for (int v = tid; v < DV; v += 256) lmax = fmaxf(lmax, yr[v]);
  s_red[tid] = lmax; __syncthreads();
  for (int off = 128; off > 0; off >>= 1) {
    if (tid < off) s_red[tid] = fmaxf(s_red[tid], s_red[tid + off]);
    __syncthreads();
  }
  float rmax = s_red[0]; __syncthreads();
  float lsum = 0.0f;
  for (int v = tid; v < DV; v += 256) lsum += __expf(yr[v] - rmax);
  s_red[tid] = lsum; __syncthreads();
  for (int off = 128; off > 0; off >>= 1) {
    if (tid < off) s_red[tid] += s_red[tid + off];
    __syncthreads();
  }
  float gscale = g[row] * fast_rcp(s_red[0]);
  for (int v = tid; v < DV; v += 256) yr[v] = __expf(yr[v] - rmax) * gscale;
  for (int v = DV + tid; v < DV + EXT; v += 256) yr[v] = 0.0f;
}

// ---------------- pointer scatter-add ----------------
__global__ void scatter_kernel(const int* __restrict__ x_index, const float* __restrict__ waws,
                               const float* __restrict__ g, float* __restrict__ Y) {
  int i = blockIdx.x * blockDim.x + threadIdx.x;
  if (i >= TT * BB * SS) return;
  int s = i % SS;
  int b = (i / SS) % BB;
  int t = i / (SS * BB);
  int row = t * BB + b;
  int v = x_index[s * BB + b];
  float add = (1.0f - g[row]) * waws[(size_t)row * SS + s];
  atomicAdd(Y + (size_t)row * (DV + EXT) + v, add);
}

// =====================================================================
extern "C" void kernel_launch(void* const* d_in, const int* in_sizes, int n_in,
                              void* d_out, int out_size, void* d_ws, size_t ws_size,
                              hipStream_t stream) {
  const float* emb      = (const float*)d_in[0];   // [T,B,E]
  const float* enc      = (const float*)d_in[1];   // [S,B,2H]
  const float* init_st  = (const float*)d_in[2];   // [B,H]
  const float* xmask    = (const float*)d_in[3];   // [S,B,1]
  const float* ymask    = (const float*)d_in[4];   // [T,B,1]
  const int*   x_index  = (const int*)d_in[5];     // [S,B]
  const float* init_cov = (const float*)d_in[7];   // [B,S]
  const float* W_h      = (const float*)d_in[8];
  const float* W_s      = (const float*)d_in[9];
  const float* W_c      = (const float*)d_in[10];
  const float* b_attn   = (const float*)d_in[11];
  const float* V_attn   = (const float*)d_in[12];
  const float* g1_Wih   = (const float*)d_in[13];
  const float* g1_Whh   = (const float*)d_in[14];
  const float* g1_bih   = (const float*)d_in[15];
  const float* g1_bhh   = (const float*)d_in[16];
  const float* g2_Wih   = (const float*)d_in[17];
  const float* g2_Whh   = (const float*)d_in[18];
  const float* g2_bih   = (const float*)d_in[19];
  const float* g2_bhh   = (const float*)d_in[20];
  const float* V1_w     = (const float*)d_in[21];
  const float* V1_b     = (const float*)d_in[22];
  const float* V2_w     = (const float*)d_in[23];
  const float* V2_b     = (const float*)d_in[24];
  const float* Wp_w     = (const float*)d_in[25];
  const float* Wp_b     = (const float*)d_in[26];

  float* out = (float*)d_out;
  const size_t YP_OFF = 0;
  const size_t HS_OFF  = (size_t)TT * BB * (DV + EXT);           // 51,251,200
  const size_t WAW_OFF = HS_OFF + (size_t)TT * BB * HH;
  const size_t COV_OFF = WAW_OFF + (size_t)TT * BB * SS;

  // ---- workspace carve-up ----
  char* wsp = (char*)d_ws;
  size_t off = 0;
  auto alloc = [&](size_t bytes) -> void* {
    void* p = wsp + off;
    off += (bytes + 255) & ~(size_t)255;
    return p;
  };
  float*  enc_proj = (float*)alloc((size_t)SS * BB * C2 * 4);   // b_attn folded in
  float*  ctx_all  = (float*)alloc((size_t)TB * C2 * 4);
  float*  hcat     = (float*)alloc((size_t)TB * 896 * 4);
  float*  tmp1     = (float*)alloc((size_t)TB * HH * 4);
  __bf16* tmp1bf   = (__bf16*)alloc((size_t)TB * HH * 2);
  __bf16* V2bf     = (__bf16*)alloc((size_t)DV * HH * 2);
  float*  gxA      = (float*)alloc((size_t)BB * 3 * HH * 4);
  float*  ghA      = (float*)alloc((size_t)BB * 3 * HH * 4);
  float*  s1       = (float*)alloc((size_t)BB * HH * 4);
  float*  s1W      = (float*)alloc((size_t)BB * C2 * 4);
  float*  hstate   = (float*)alloc((size_t)BB * HH * 4);
  float*  cov      = (float*)alloc((size_t)BB * SS * 4);
  float*  score    = (float*)alloc((size_t)SS * BB * 4);
  float*  gbuf     = (float*)alloc((size_t)TB * 4);

  // ---- state init ----
  hipMemcpyAsync(hstate, init_st, (size_t)BB * HH * 4, hipMemcpyDeviceToDevice, stream);
  hipMemcpyAsync(cov, init_cov, (size_t)BB * SS * 4, hipMemcpyDeviceToDevice, stream);

  // ---- one-time precompute ----
  // enc_proj[S*B,512] = enc @ W_h^T + b_attn   (M=6400, N=512, K=512)
  gemm_xwT_f32<<<dim3(512 / 16 / 4, SS * BB / 16), 128, 0, stream>>>(
      enc, W_h, b_attn, enc_proj, SS * BB, C2, C2, 0);
  // V2_w -> bf16 (12.8M elems; stays resident in L2 for the vocab GEMM)
  {
    size_t n = (size_t)DV * HH;
    f32_to_bf16_kernel<<<(unsigned)((n + 255) / 256), 256, 0, stream>>>(V2_w, V2bf, n);
  }

  // ---- sequential decoder scan ----
  for (int t = 0; t < TT; ++t) {
    const float* emb_t = emb + (size_t)t * BB * EE;
    const float* ym_t  = ymask + (size_t)t * BB;
    float* ctx_t  = ctx_all + (size_t)t * BB * C2;
    float* waws_t = out + WAW_OFF + (size_t)t * BB * SS;
    float* covs_t = out + COV_OFF + (size_t)t * BB * SS;
    float* hs_t   = out + HS_OFF + (size_t)t * BB * HH;

    // GRU1 gates
    gemm_xwT_f32<<<dim3(12, 1), 128, 0, stream>>>(emb_t, g1_Wih, g1_bih, gxA, BB, 3 * HH, EE, 0);
    gemm_xwT_f32<<<dim3(12, 1), 128, 0, stream>>>(hstate, g1_Whh, g1_bhh, ghA, BB, 3 * HH, HH, 0);
    gru_combine_kernel<<<(BB * HH + 255) / 256, 256, 0, stream>>>(gxA, ghA, hstate, ym_t, s1, nullptr);

    // attention
    gemm_xwT_f32<<<dim3(8, 1), 128, 0, stream>>>(s1, W_s, nullptr, s1W, BB, C2, HH, 0);
    attn_score_kernel<<<SS * BB / 4, 128, 0, stream>>>(enc_proj, s1W, W_c, V_attn, cov, xmask, score);
    attn_ctx_kernel<<<BB, 256, 0, stream>>>(score, enc, cov, ctx_t, waws_t, covs_t);

    // GRU2 gates
    gemm_xwT_f32<<<dim3(12, 1), 128, 0, stream>>>(ctx_t, g2_Wih, g2_bih, gxA, BB, 3 * HH, C2, 0);
    gemm_xwT_f32<<<dim3(12, 1), 128, 0, stream>>>(s1, g2_Whh, g2_bhh, ghA, BB, 3 * HH, HH, 0);
    gru_combine_kernel<<<(BB * HH + 255) / 256, 256, 0, stream>>>(gxA, ghA, s1, ym_t, hstate, hs_t);
  }

  // ---- word-prob layer ----
  pack_hcat_kernel<<<(TB * 896 + 255) / 256, 256, 0, stream>>>(out + HS_OFF, ctx_all, emb, hcat);
  // tmp1 = tanh(hcat @ V1_w^T + V1_b)   (M=1024, N=256, K=896)
  gemm_xwT_f32<<<dim3(HH / 16 / 4, TB / 16), 128, 0, stream>>>(hcat, V1_w, V1_b, tmp1, TB, HH, 896, 1);
  f32_to_bf16_kernel<<<(TB * HH + 255) / 256, 256, 0, stream>>>(tmp1, tmp1bf, (size_t)TB * HH);
  gate_kernel<<<TB / 4, 128, 0, stream>>>(hcat, Wp_w, Wp_b, gbuf);

  // logits straight into d_out (row stride DV+EXT); then in-place softmax*g
  gemm_vocab_bf16<<<dim3((DV / 16 + 3) / 4, 1), 128, 0, stream>>>(
      tmp1bf, V2bf, V2_b, out + YP_OFF, TB, DV, HH, DV + EXT);
  vocab_softmax_kernel<<<TB, 256, 0, stream>>>(out + YP_OFF, gbuf);

  // pointer distribution scatter-add
  scatter_kernel<<<(TT * BB * SS + 255) / 256, 256, 0, stream>>>(
      x_index, out + WAW_OFF, gbuf, out + YP_OFF);
}
